// BTD_AutoCorrelationLayer_19198503813441
// MI455X (gfx1250) — matrix operand order repaired and checked by reading, not compiled, not run on hardware
//
#include <hip/hip_runtime.h>

// ---------------------------------------------------------------------------
// BTD AutoCorrelation layer, restructured as:
//   Qp/Kp/Vp = X @ W + b                  (bf16 WMMA GEMMs, f32 accum)
//   G[n,b]   = U[n,b] @ Wo                (dominant: 64 x 9216 x 512, x256)
//              U[n,b][h, j*96+k] = Kp[nb][j,h] * Vp[nb][k,h]  (built in LDS)
//   out[b]   = (1/8) * sum_n (core[n]*Qp[nb]) @ G[nb] + bo
// All matrix math via v_wmma_f32_16x16x32_bf16 (wave32).
// B-side operands stored K-contiguous so fragments load as 2 x b128.
// K/V slices staged to LDS via GLOBAL_LOAD_ASYNC_TO_LDS_B128 (ASYNCcnt).
// ---------------------------------------------------------------------------

typedef __attribute__((ext_vector_type(16))) __bf16         v16bf;
typedef __attribute__((ext_vector_type(16))) unsigned short ushort16;
typedef __attribute__((ext_vector_type(8)))  unsigned short ushort8;
typedef __attribute__((ext_vector_type(8)))  float          v8f;
typedef __attribute__((ext_vector_type(4)))  int            v4i;

#if defined(__has_builtin)
#if __has_builtin(__builtin_amdgcn_global_load_async_to_lds_b128)
#define HAVE_ASYNC_LDS 1
#endif
#endif

typedef __attribute__((address_space(1))) v4i* gv4i_p;
typedef __attribute__((address_space(3))) v4i* lv4i_p;

__device__ __forceinline__ unsigned short f2bf(float f) {
  unsigned int u = __builtin_bit_cast(unsigned int, f);
  u += 0x7FFFu + ((u >> 16) & 1u);   // round-to-nearest-even
  return (unsigned short)(u >> 16);
}

__device__ __forceinline__ v16bf frag_from_chunks(ushort8 lo, ushort8 hi) {
  ushort16 s = __builtin_shufflevector(lo, hi, 0, 1, 2, 3, 4, 5, 6, 7,
                                       8, 9, 10, 11, 12, 13, 14, 15);
  return __builtin_bit_cast(v16bf, s);
}

// A-matrix fragment: 16x32 (MxK) bf16 tile, row-major, row stride `stride`.
// Lane needs row m = lane&15, K range [hf*16, hf*16+16) (matches B frag)
// -> two contiguous 16B loads.
__device__ __forceinline__ v16bf load_a_frag(const unsigned short* a, int stride) {
  const int lane = threadIdx.x & 31;
  const int m  = lane & 15;
  const int hf = lane >> 4;
  const unsigned short* p = a + m * stride;
  ushort8 lo = *(const ushort8*)(p + hf * 16);
  ushort8 hi = *(const ushort8*)(p + hf * 16 + 8);
  return frag_from_chunks(lo, hi);
}

// B-matrix fragment from K-contiguous storage bt[n][k] (row stride `stride`):
// lane needs B[kb..kb+16)[n] = bt[n][kb..kb+16) -> two contiguous 16B loads.
__device__ __forceinline__ v16bf load_b_frag_t(const unsigned short* bt, int stride) {
  const int lane = threadIdx.x & 31;
  const int n  = lane & 15;
  const int kb = (lane >> 4) * 16;
  const unsigned short* p = bt + n * stride + kb;
  ushort8 lo = *(const ushort8*)(p);
  ushort8 hi = *(const ushort8*)(p + 8);
  return frag_from_chunks(lo, hi);
}

// ---------------------------------------------------------------------------
__global__ void cvt_f32_bf16(const float* __restrict__ src,
                             unsigned short* __restrict__ dst, int n) {
  const int i = blockIdx.x * blockDim.x + threadIdx.x;
  if (i < n) dst[i] = f2bf(src[i]);
}

// Transpose-convert a 512x512 f32 weight W[k][n] into bf16 Wt[n][k].
__global__ void cvt_xpose_512(const float* __restrict__ src,
                              unsigned short* __restrict__ dst) {
  const int id = blockIdx.x * 256 + threadIdx.x;  // 0..262143, k fastest
  const int k = id & 511;
  const int n = id >> 9;
  dst[n * 512 + k] = f2bf(src[k * 512 + n]);
}

// ---------------------------------------------------------------------------
// C[3072,512] = Xbf[3072,512] @ W + bias, with W given K-contiguous (Wt[n][k]).
// One wave per block; each wave: 1 M-tile x 2 N-tiles (A reused twice).
__global__ void proj_gemm(const unsigned short* __restrict__ X,
                          const unsigned short* __restrict__ Wt,
                          const float* __restrict__ bias,
                          float* __restrict__ out) {
  const int nt2 = blockIdx.x;  // 0..15 (pairs of N tiles)
  const int mt  = blockIdx.y;  // 0..191
  const int lane = threadIdx.x & 31;
  v8f acc0 = {}, acc1 = {};
#pragma unroll 4
  for (int k0 = 0; k0 < 512; k0 += 32) {
    v16bf a  = load_a_frag(X + (mt * 16) * 512 + k0, 512);
    v16bf b0 = load_b_frag_t(Wt + (nt2 * 32) * 512 + k0, 512);
    v16bf b1 = load_b_frag_t(Wt + (nt2 * 32 + 16) * 512 + k0, 512);
    acc0 = __builtin_amdgcn_wmma_f32_16x16x32_bf16(false, a, false, b0,
                                                   (short)0, acc0, false, false);
    acc1 = __builtin_amdgcn_wmma_f32_16x16x32_bf16(false, a, false, b1,
                                                   (short)0, acc1, false, false);
  }
  const int hf  = lane >> 4;
  const int c0  = nt2 * 32 + (lane & 15);
  const int c1  = c0 + 16;
  const float bb0 = bias[c0], bb1 = bias[c1];
#pragma unroll
  for (int r = 0; r < 8; ++r) {
    const int row = mt * 16 + r + 8 * hf;
    out[row * 512 + c0] = acc0[r] + bb0;
    out[row * 512 + c1] = acc1[r] + bb1;
  }
}

// ---------------------------------------------------------------------------
// Per (n,b): G = U[64,9216] @ Wo[9216,512], U built on the fly in LDS.
// Block: 256 threads = 8 waves as 4(M) x 2(N-half); block tile M=64, N=64.
// Output stored transposed: Gt[nb][d][h] so the next GEMM's B-frags are
// contiguous, and each lane's 8 results become one 16B store.
__global__ void kv_outer_gemm(const float* __restrict__ Kp,
                              const float* __restrict__ Vp,
                              const unsigned short* __restrict__ Wo,
                              unsigned short* __restrict__ Gt) {
  __shared__ float          s_k[96 * 64];    // 24 KB
  __shared__ float          s_v[96 * 64];    // 24 KB
  __shared__ unsigned short s_u[64 * 32];    //  4 KB  U K-tile [h][k]
  __shared__ unsigned short s_wt[64 * 32];   //  4 KB  Wo K-tile transposed [n][k]

  const int tid = threadIdx.x;
  const int nb  = blockIdx.z;                // n*32 + b (reshape trick)
  const int nb0 = blockIdx.x * 64;           // N base of this block

#if HAVE_ASYNC_LDS
  {
    gv4i_p gk = (gv4i_p)(unsigned long long)(Kp + nb * 6144);
    gv4i_p gv = (gv4i_p)(unsigned long long)(Vp + nb * 6144);
    lv4i_p lk = (lv4i_p)(unsigned int)(unsigned long long)(void*)s_k;
    lv4i_p lv = (lv4i_p)(unsigned int)(unsigned long long)(void*)s_v;
    for (int i = tid; i < 1536; i += 256) {   // 1536 x 16B = 24 KB each
      __builtin_amdgcn_global_load_async_to_lds_b128(gk + i, lk + i, 0, 0);
      __builtin_amdgcn_global_load_async_to_lds_b128(gv + i, lv + i, 0, 0);
    }
#if __has_builtin(__builtin_amdgcn_s_wait_asynccnt)
    __builtin_amdgcn_s_wait_asynccnt(0);
#else
    asm volatile("s_wait_asynccnt 0" ::: "memory");
#endif
  }
#else
  for (int i = tid; i < 6144; i += 256) {
    s_k[i] = Kp[nb * 6144 + i];              // [j][h], h contiguous
    s_v[i] = Vp[nb * 6144 + i];
  }
#endif
  __syncthreads();

  const int wave = tid >> 5;
  const int lane = tid & 31;
  const int mi   = wave & 3;                 // M tile (rank rows)
  const int ni   = wave >> 2;                // which 32-col half of block N

  const int uh  = tid >> 2;                  // 0..63 (rank row)
  const int uc0 = (tid & 3) * 8;
  const int kq  = tid >> 3;                  // 0..31 (Wo k row)
  const int nl0 = (tid & 7) * 8;

  v8f acc0 = {}, acc1 = {};
  for (int t = 0; t < 288; ++t) {
    // Build U K-tile: U[h, c] = k[j,h]*v[kc,h], jk = t*32+c
#pragma unroll
    for (int i = 0; i < 8; ++i) {
      const int c  = uc0 + i;
      const int jk = t * 32 + c;
      const int j  = jk / 96;
      const int kc = jk - j * 96;
      s_u[uh * 32 + c] = f2bf(s_k[j * 64 + uh] * s_v[kc * 64 + uh]);
    }
    // Stage Wo K-tile transposed: s_wt[nl][k] = Wo[t*32+k][nb0+nl]
#pragma unroll
    for (int i = 0; i < 8; ++i) {
      const int nl = nl0 + i;
      s_wt[nl * 32 + kq] = Wo[(t * 32 + kq) * 512 + nb0 + nl];
    }
    // Prefetch next Wo K-tile (one row per low lane)
    if (t + 1 < 288 && tid < 32)
      __builtin_prefetch(Wo + ((t + 1) * 32 + tid) * 512 + nb0, 0, 1);
    __syncthreads();
    v16bf a  = load_a_frag(s_u + mi * 16 * 32, 32);
    v16bf b0 = load_b_frag_t(s_wt + (ni * 32) * 32, 32);
    v16bf b1 = load_b_frag_t(s_wt + (ni * 32 + 16) * 32, 32);
    acc0 = __builtin_amdgcn_wmma_f32_16x16x32_bf16(false, a, false, b0,
                                                   (short)0, acc0, false, false);
    acc1 = __builtin_amdgcn_wmma_f32_16x16x32_bf16(false, a, false, b1,
                                                   (short)0, acc1, false, false);
    __syncthreads();
  }
  // Store transposed: Gt[nb*32768 + col*64 + h]; 8 h-values contiguous.
  const int hf   = lane >> 4;
  const int hrow = mi * 16 + hf * 8;
  const int c0   = nb0 + ni * 32 + (lane & 15);
  const int c1   = c0 + 16;
  ushort8 p0, p1;
#pragma unroll
  for (int r = 0; r < 8; ++r) { p0[r] = f2bf(acc0[r]); p1[r] = f2bf(acc1[r]); }
  *(ushort8*)(Gt + nb * 32768 + c0 * 64 + hrow) = p0;
  *(ushort8*)(Gt + nb * 32768 + c1 * 64 + hrow) = p1;
}

// ---------------------------------------------------------------------------
// out[b] = (1/8) * sum_n (core[n] * Qp[nb]) @ G[nb] + bo
// Block: 256 threads = 8 waves as 2(M) x 4(N); block tile M=32, N=128.
// B-frags read straight from Gt (K-contiguous) as 2 x b128 global loads.
__global__ void qg_gemm(const float* __restrict__ Qp,
                        const float* __restrict__ core,
                        const unsigned short* __restrict__ Gt,
                        const float* __restrict__ bo,
                        float* __restrict__ out) {
  __shared__ unsigned short s_a[32 * 32];    // q' K-tile, bf16

  const int tid   = threadIdx.x;
  const int b     = blockIdx.z;              // batch
  const int mbase = blockIdx.y * 32;         // row base (0/32/64)
  const int wave  = tid >> 5;
  const int lane  = tid & 31;
  const int mi    = wave & 1;
  const int ni    = wave >> 1;               // 0..3
  const int ncb   = blockIdx.x * 128 + ni * 32;

  v8f acc0 = {}, acc1 = {};
  for (int n = 0; n < 8; ++n) {
    const int nb = n * 32 + b;
#pragma unroll
    for (int kk0 = 0; kk0 < 64; kk0 += 32) {
#pragma unroll
      for (int i = 0; i < 4; ++i) {
        const int id = tid * 4 + i;          // = il*32 + hc
        const int il = id >> 5;
        const int hc = id & 31;
        const float q = Qp[nb * 6144 + (mbase + il) * 64 + kk0 + hc];
        s_a[id] = f2bf(core[n * 64 + kk0 + hc] * q);
      }
      __syncthreads();
      v16bf a  = load_a_frag(s_a + mi * 16 * 32, 32);
      v16bf b0 = load_b_frag_t(Gt + nb * 32768 + ncb * 64 + kk0, 64);
      v16bf b1 = load_b_frag_t(Gt + nb * 32768 + (ncb + 16) * 64 + kk0, 64);
      acc0 = __builtin_amdgcn_wmma_f32_16x16x32_bf16(false, a, false, b0,
                                                     (short)0, acc0, false, false);
      acc1 = __builtin_amdgcn_wmma_f32_16x16x32_bf16(false, a, false, b1,
                                                     (short)0, acc1, false, false);
      __syncthreads();
    }
  }
  const int hf = lane >> 4;
  const int c0 = ncb + (lane & 15);
  const int c1 = c0 + 16;
  const float bb0 = bo[c0], bb1 = bo[c1];
#pragma unroll
  for (int r = 0; r < 8; ++r) {
    const int row = mbase + mi * 16 + r + 8 * hf;
    out[b * 96 * 512 + row * 512 + c0] = acc0[r] * 0.125f + bb0;
    out[b * 96 * 512 + row * 512 + c1] = acc1[r] * 0.125f + bb1;
  }
}

// ---------------------------------------------------------------------------
extern "C" void kernel_launch(void* const* d_in, const int* in_sizes, int n_in,
                              void* d_out, int out_size, void* d_ws, size_t ws_size,
                              hipStream_t stream) {
  (void)in_sizes; (void)n_in; (void)out_size; (void)ws_size;
  const float* queries = (const float*)d_in[0];
  const float* keys    = (const float*)d_in[1];
  const float* values  = (const float*)d_in[2];
  // d_in[3] = attn_mask (all false, unused by reference math)
  const float* Wq   = (const float*)d_in[4];
  const float* bq   = (const float*)d_in[5];
  const float* Wk   = (const float*)d_in[6];
  const float* bk   = (const float*)d_in[7];
  const float* Wv   = (const float*)d_in[8];
  const float* bv   = (const float*)d_in[9];
  const float* core = (const float*)d_in[10];
  const float* Wo   = (const float*)d_in[11];
  const float* bo   = (const float*)d_in[12];
  float* out = (float*)d_out;

  const int NE  = 32 * 96 * 512;   // 1,572,864  (B*L*D)
  const int NW  = 512 * 512;       //   262,144
  const int NWO = 9216 * 512;      // 4,718,592
  const int NG  = 256 * 64 * 512;  // 8,388,608

  char* ws = (char*)d_ws;
  size_t off = 0;
  auto alloc = [&](size_t bytes) -> char* {
    char* p = ws + off;
    off = (off + bytes + 255) & ~size_t(255);
    return p;
  };
  unsigned short* Qinbf = (unsigned short*)alloc((size_t)NE * 2);
  unsigned short* Kinbf = (unsigned short*)alloc((size_t)NE * 2);
  unsigned short* Vinbf = (unsigned short*)alloc((size_t)NE * 2);
  unsigned short* Wqt   = (unsigned short*)alloc((size_t)NW * 2);
  unsigned short* Wkt   = (unsigned short*)alloc((size_t)NW * 2);
  unsigned short* Wvt   = (unsigned short*)alloc((size_t)NW * 2);
  unsigned short* Wobf  = (unsigned short*)alloc((size_t)NWO * 2);
  float*          Qp    = (float*)alloc((size_t)NE * 4);
  float*          Kp    = (float*)alloc((size_t)NE * 4);
  float*          Vp    = (float*)alloc((size_t)NE * 4);
  unsigned short* Gt    = (unsigned short*)alloc((size_t)NG * 2);

  // 1) f32 -> bf16 conversions (weights transposed to K-contiguous)
  cvt_f32_bf16<<<(NE  + 255) / 256, 256, 0, stream>>>(queries, Qinbf, NE);
  cvt_f32_bf16<<<(NE  + 255) / 256, 256, 0, stream>>>(keys,    Kinbf, NE);
  cvt_f32_bf16<<<(NE  + 255) / 256, 256, 0, stream>>>(values,  Vinbf, NE);
  cvt_xpose_512<<<1024, 256, 0, stream>>>(Wq, Wqt);
  cvt_xpose_512<<<1024, 256, 0, stream>>>(Wk, Wkt);
  cvt_xpose_512<<<1024, 256, 0, stream>>>(Wv, Wvt);
  cvt_f32_bf16<<<(NWO + 255) / 256, 256, 0, stream>>>(Wo, Wobf, NWO);

  // 2) projections (3072x512 @ 512x512 + bias)
  proj_gemm<<<dim3(16, 192), 32, 0, stream>>>(Qinbf, Wqt, bq, Qp);
  proj_gemm<<<dim3(16, 192), 32, 0, stream>>>(Kinbf, Wkt, bk, Kp);
  proj_gemm<<<dim3(16, 192), 32, 0, stream>>>(Vinbf, Wvt, bv, Vp);

  // 3) Gt[nb] = (U[nb] @ Wo)^T   (dominant GEMM)
  kv_outer_gemm<<<dim3(8, 1, 256), 256, 0, stream>>>(Kp, Vp, Wobf, Gt);

  // 4) out = (1/8) * sum_n q'[nb] @ G[nb] + bo
  qg_gemm<<<dim3(4, 3, 32), 256, 0, stream>>>(Qp, core, Gt, bo, out);
}
